// QGNN_80401787781121
// MI455X (gfx1250) — matrix operand order repaired and checked by reading, not compiled, not run on hardware
//
#include <hip/hip_runtime.h>

#define NV_    6400
#define NNODES 100
#define BATCH  64
#define EF     128
#define DEG    99

typedef __attribute__((ext_vector_type(16))) _Float16 v16h;
typedef __attribute__((ext_vector_type(8)))  float    v8f;

union Frag16 { v16h v; _Float16 h[16]; };

// ---------------------------------------------------------------------------
// Prep: rank-2 collapse vectors u± = relu(±p) @ Ww for both layers,
//       fused layer-1 input weights Wxf1 = Wx1 + 99*Wf1,
//       and transposed pre-scaled f16 B operand  Wf2t[n,k] = 99*Wf2[k,n].
// ---------------------------------------------------------------------------
__global__ void prep_kernel(const float* __restrict__ Wx1, const float* __restrict__ Ww1,
                            const float* __restrict__ Wf1, const float* __restrict__ p1,
                            const float* __restrict__ Ww2, const float* __restrict__ p2,
                            const float* __restrict__ Wf2,
                            float* upos1, float* uneg1, float* upos2, float* uneg2,
                            float* Wxf1, _Float16* Wf2t)
{
    int k = threadIdx.x;                       // 0..127
    float up1 = 0.f, un1 = 0.f, up2 = 0.f, un2 = 0.f;
    for (int j = 0; j < EF; ++j) {
        float w1 = Ww1[j * EF + k], w2 = Ww2[j * EF + k];
        float a = p1[j], b = p2[j];
        up1 += fmaxf(a, 0.f) * w1;  un1 += fminf(a, 0.f) * w1;
        up2 += fmaxf(b, 0.f) * w2;  un2 += fminf(b, 0.f) * w2;
    }
    upos1[k] = up1; uneg1[k] = un1; upos2[k] = up2; uneg2[k] = un2;
    Wxf1[k]      = Wx1[k]      + 99.f * Wf1[k];
    Wxf1[EF + k] = Wx1[EF + k] + 99.f * Wf1[EF + k];
    for (int j = 0; j < EF; ++j)
        Wf2t[k * EF + j] = (_Float16)(99.f * Wf2[j * EF + k]);
}

// ---------------------------------------------------------------------------
// Edge reduction: one wave32 per node; 99 contiguous edge weights per node.
// Spos = sum of max(w,0), Sneg = sum of min(w,0).
// ---------------------------------------------------------------------------
__global__ void edge_kernel(const float* __restrict__ w, float* Spos, float* Sneg)
{
    int wave = (blockIdx.x * blockDim.x + threadIdx.x) >> 5;
    int lane = threadIdx.x & 31;
    if (wave >= NV_) return;
    long base = (long)wave * DEG;
    float v0 = w[base + lane], v1 = w[base + 32 + lane], v2 = w[base + 64 + lane];
    float p = fmaxf(v0, 0.f) + fmaxf(v1, 0.f) + fmaxf(v2, 0.f);
    float n = fminf(v0, 0.f) + fminf(v1, 0.f) + fminf(v2, 0.f);
    if (lane < 3) {
        float v3 = w[base + 96 + lane];
        p += fmaxf(v3, 0.f); n += fminf(v3, 0.f);
    }
    for (int off = 16; off > 0; off >>= 1) {
        p += __shfl_xor(p, off, 32);
        n += __shfl_xor(n, off, 32);
    }
    if (lane == 0) { Spos[wave] = p; Sneg[wave] = n; }
}

// ---------------------------------------------------------------------------
// Layer 1 (fully collapsed): h1[d,k] = relu(x[d]@Wxf1[:,k]
//                                    + Spos[d]*u+1[k] + Sneg[d]*u-1[k] + bf1[k])
// stored directly as f16 (it is only consumed as the WMMA A operand).
// ---------------------------------------------------------------------------
__global__ void layer1_kernel(const float* __restrict__ x,
                              const float* __restrict__ Spos, const float* __restrict__ Sneg,
                              const float* __restrict__ upos1, const float* __restrict__ uneg1,
                              const float* __restrict__ Wxf1, const float* __restrict__ bf1,
                              _Float16* __restrict__ h1)
{
    int idx = blockIdx.x * blockDim.x + threadIdx.x;
    if (idx >= NV_ * EF) return;
    int d = idx >> 7, k = idx & (EF - 1);
    float v = x[2 * d] * Wxf1[k] + x[2 * d + 1] * Wxf1[EF + k]
            + Spos[d] * upos1[k] + Sneg[d] * uneg1[k] + bf1[k];
    h1[idx] = (_Float16)fmaxf(v, 0.f);
}

// ---------------------------------------------------------------------------
// Layer 2 GEMM via WMMA: h2 = relu( (99*h1) @ Wf2  + x@Wx2 + Spos*u+2 + Sneg*u-2 + bf2 )
// One wave per 16x16 tile; K=128 as 4 x v_wmma_f32_16x16x32_f16.
// ---------------------------------------------------------------------------
__global__ void __launch_bounds__(256) gemm2_kernel(
    const _Float16* __restrict__ h1, const _Float16* __restrict__ Bt,
    const float* __restrict__ x,
    const float* __restrict__ Spos, const float* __restrict__ Sneg,
    const float* __restrict__ Wx2,
    const float* __restrict__ upos2, const float* __restrict__ uneg2,
    const float* __restrict__ bf2, float* __restrict__ h2)
{
    const int waveN = threadIdx.x >> 5;          // 8 waves -> 8 N-tiles (N=128)
    const int lane  = threadIdx.x & 31;
    const int m16   = lane & 15;
    const int hi    = lane >> 4;
    const int Mbase = blockIdx.x * 16;
    const int Nbase = waveN * 16;

    v8f acc = {};
    const _Float16* Arow = h1 + (long)(Mbase + m16) * EF;   // A row-major (M x K)
    const _Float16* Brow = Bt + (long)(Nbase + m16) * EF;   // B stored K-contiguous per column

#pragma unroll
    for (int kt = 0; kt < 4; ++kt) {
        const int kb = kt * 32;
        Frag16 a, b;
        // 16-bit A 16x32 layout: lanes 0-15 halves 0..7 -> K 0..7, 8..15 -> K 16..23;
        // lanes 16-31 shifted by +8.
        *(uint4*)&a.h[0] = *(const uint4*)(Arow + kb + hi * 8);
        *(uint4*)&a.h[8] = *(const uint4*)(Arow + kb + hi * 8 + 16);
        // B: lane holds column n, halves 0..7 -> K 8*hi+0..7, 8..15 -> K 16+8*hi+0..7 (same as A).
        *(uint4*)&b.h[0] = *(const uint4*)(Brow + kb + hi * 8);
        *(uint4*)&b.h[8] = *(const uint4*)(Brow + kb + hi * 8 + 16);
        acc = __builtin_amdgcn_wmma_f32_16x16x32_f16(
            /*neg_a=*/false, a.v, /*neg_b=*/false, b.v,
            /*c_mod=*/(short)0, acc, /*reuse_a=*/false, /*reuse_b=*/false);
    }

    // Fused epilogue: C/D layout -> VGPR r = row (Mbase + r + 8*hi), col (Nbase + m16)
    const int n = Nbase + m16;
    const float wx0 = Wx2[n], wx1 = Wx2[EF + n];
    const float up = upos2[n], un = uneg2[n], bb = bf2[n];
#pragma unroll
    for (int r = 0; r < 8; ++r) {
        int d = Mbase + r + hi * 8;
        float v = acc[r] + x[2 * d] * wx0 + x[2 * d + 1] * wx1
                + Spos[d] * up + Sneg[d] * un + bb;
        h2[(long)d * EF + n] = fmaxf(v, 0.f);
    }
}

// ---------------------------------------------------------------------------
// Readout: per graph -> sum-pool@W6, gather-v@W7, action@W8, relu, concat@W5.
// ---------------------------------------------------------------------------
__global__ void readout_kernel(const float* __restrict__ h2, const float* __restrict__ action,
                               const int* __restrict__ v,
                               const float* __restrict__ W5, const float* __restrict__ b5,
                               const float* __restrict__ W6, const float* __restrict__ b6,
                               const float* __restrict__ W7, const float* __restrict__ b7,
                               const float* __restrict__ W8, const float* __restrict__ b8,
                               float* __restrict__ out)
{
    __shared__ float s[EF], cur[EF], part[EF];
    const int b = blockIdx.x, k = threadIdx.x;
    const float* hb = h2 + (long)b * NNODES * EF;

    float acc = 0.f;
    for (int i = 0; i < NNODES; ++i) acc += hb[i * EF + k];
    s[k]   = acc;
    cur[k] = hb[v[b] * EF + k];
    __syncthreads();

    float a1 = b6[k], a2 = b7[k];
    for (int j = 0; j < EF; ++j) {
        a1 += s[j]   * W6[j * EF + k];
        a2 += cur[j] * W7[j * EF + k];
    }
    float a3 = action[b] * W8[k] + b8[k];
    part[k] = fmaxf(a1, 0.f) * W5[k]
            + fmaxf(a2, 0.f) * W5[EF + k]
            + fmaxf(a3, 0.f) * W5[2 * EF + k];
    __syncthreads();
    for (int off = 64; off > 0; off >>= 1) {
        if (k < off) part[k] += part[k + off];
        __syncthreads();
    }
    if (k == 0) out[b] = part[0] + b5[0];
}

// ---------------------------------------------------------------------------
extern "C" void kernel_launch(void* const* d_in, const int* in_sizes, int n_in,
                              void* d_out, int out_size, void* d_ws, size_t ws_size,
                              hipStream_t stream)
{
    (void)in_sizes; (void)n_in; (void)out_size; (void)ws_size;

    const float* x      = (const float*)d_in[0];
    const float* w      = (const float*)d_in[1];
    const float* action = (const float*)d_in[2];
    /* d_in[3] dst: complete-graph structure is known (contiguous 99-edge
       segments per node) so dst is not needed on device. */
    const int*   v      = (const int*)d_in[4];
    const float* Wx1 = (const float*)d_in[5];
    const float* Ww1 = (const float*)d_in[6];
    const float* Wf1 = (const float*)d_in[7];
    const float* bf1 = (const float*)d_in[8];
    const float* p1  = (const float*)d_in[9];
    const float* Wx2 = (const float*)d_in[10];
    const float* Ww2 = (const float*)d_in[11];
    const float* Wf2 = (const float*)d_in[12];
    const float* bf2 = (const float*)d_in[13];
    const float* p2  = (const float*)d_in[14];
    const float* W5  = (const float*)d_in[15];
    const float* b5  = (const float*)d_in[16];
    const float* W6  = (const float*)d_in[17];
    const float* b6  = (const float*)d_in[18];
    const float* W7  = (const float*)d_in[19];
    const float* b7  = (const float*)d_in[20];
    const float* W8  = (const float*)d_in[21];
    const float* b8  = (const float*)d_in[22];

    char* ws = (char*)d_ws;                       // ~5.0 MB used
    float*    Spos  = (float*)(ws + 0);           // 6400 f
    float*    Sneg  = (float*)(ws + 25600);       // 6400 f
    float*    upos1 = (float*)(ws + 51200);       // 128 f
    float*    uneg1 = (float*)(ws + 51712);
    float*    upos2 = (float*)(ws + 52224);
    float*    uneg2 = (float*)(ws + 52736);
    float*    Wxf1  = (float*)(ws + 53248);       // 2x128 f
    _Float16* Wf2t  = (_Float16*)(ws + 54272);    // 128x128 f16 (16B aligned)
    _Float16* h1    = (_Float16*)(ws + 87040);    // 6400x128 f16
    float*    h2    = (float*)(ws + 1725440);     // 6400x128 f32

    prep_kernel   <<<1,    128, 0, stream>>>(Wx1, Ww1, Wf1, p1, Ww2, p2, Wf2,
                                             upos1, uneg1, upos2, uneg2, Wxf1, Wf2t);
    edge_kernel   <<<800,  256, 0, stream>>>(w, Spos, Sneg);
    layer1_kernel <<<3200, 256, 0, stream>>>(x, Spos, Sneg, upos1, uneg1, Wxf1, bf1, h1);
    gemm2_kernel  <<<400,  256, 0, stream>>>(h1, Wf2t, x, Spos, Sneg, Wx2,
                                             upos2, uneg2, bf2, h2);
    readout_kernel<<<64,   128, 0, stream>>>(h2, action, v, W5, b5, W6, b6,
                                             W7, b7, W8, b8, (float*)d_out);
}
